// YoloNet_83141976916868
// MI455X (gfx1250) — matrix operand order repaired and checked by reading, not compiled, not run on hardware
//
#include <hip/hip_runtime.h>
#include <hip/hip_bf16.h>

typedef __attribute__((ext_vector_type(16))) __bf16 v16bf;
typedef __attribute__((ext_vector_type(8)))  float  v8f;

#define L_POS     32768
#define DCH       256
#define NANCH     9
#define NCLS      21     // 1 + 20
#define MTOT      576    // 192 (cls pad) + 384 (box pad)
#define BOX_OFF   192
#define NT        32     // positions per block (2-3 blocks co-resident per WGP)
#define BSTR      264    // padded LDS row stride (halves) for feature tile
#define NBIN      1024
#define TOPK      100
#define POOLSZ    512
#define CLIP_DW   4.135166556742356f
#define SCORE_TH  0.05f

__device__ __constant__ float c_WA[9] = {64.f,128.f,256.f,128.f,256.f,512.f,256.f,512.f,1024.f};

static __device__ __forceinline__ unsigned short f32_to_bf16(float f) {
    union { float f; unsigned u; } v; v.f = f;
    unsigned r = v.u + 0x7FFFu + ((v.u >> 16) & 1u);   // round-to-nearest-even
    return (unsigned short)(r >> 16);
}

union Frag16 { uint4 u[2]; v16bf v; };

// ---------------------------------------------------------------- prep ----
__global__ void prep_weights(const float* __restrict__ cls_w,
                             const float* __restrict__ box_w,
                             unsigned short* __restrict__ Wb,
                             int* __restrict__ counters) {
    int row = blockIdx.x;          // 0..575
    int k   = threadIdx.x;         // 0..255
    float v = 0.0f;
    if (row < 189)                       v = cls_w[row * DCH + k];
    else if (row >= BOX_OFF && row < BOX_OFF + 378) v = box_w[(row - BOX_OFF) * DCH + k];
    Wb[row * DCH + k] = f32_to_bf16(v);
    if (blockIdx.x == 0 && threadIdx.x < 4) counters[threadIdx.x] = 0;
}

// ---------------------------------------------------------------- main ----
extern "C" __global__ void __launch_bounds__(256, 1)
detect_main(const float* __restrict__ F,
            const unsigned short* __restrict__ Wb,
            const float* __restrict__ cls_b,
            const float* __restrict__ box_b,
            const int* __restrict__ img,
            float4* __restrict__ cand,
            int* __restrict__ counters,
            int cap) {
    extern __shared__ char smem[];
    float*          Res  = (float*)smem;                            // [576][32] fp32
    unsigned short* Bst  = (unsigned short*)(smem + MTOT * NT * 4); // [32][BSTR] bf16
    int*            hist = (int*)Bst;                               // aliased after GEMM
    __shared__ int s_cut;

    const int tid  = threadIdx.x;
    const int lane = tid & 31;
    const int wid  = tid >> 5;          // 0..7
    const int b    = blockIdx.y;
    const int l0   = blockIdx.x * NT;

    const float imsz   = (float)(*img);
    const float stride = imsz / (float)L_POS;

    // ---- stage feature tile (fp32 -> bf16, transposed [n][k]) ----
    // float4 loads over adjacent channel pairs; packed 2xbf16 b32 LDS stores
    {
        const float* fb = F + (size_t)b * ((size_t)DCH * L_POS) + l0;
        const int n0 = (tid & 7) * 4;    // 0,4,...,28
        const int kq = tid >> 3;         // 0..31
        #pragma unroll
        for (int i = 0; i < 4; ++i) {
            const int k = kq * 8 + i * 2;
            const float4 fa = *reinterpret_cast<const float4*>(fb + (size_t)k       * L_POS + n0);
            const float4 fc = *reinterpret_cast<const float4*>(fb + (size_t)(k + 1) * L_POS + n0);
            const float va[4] = {fa.x, fa.y, fa.z, fa.w};
            const float vc[4] = {fc.x, fc.y, fc.z, fc.w};
            #pragma unroll
            for (int j = 0; j < 4; ++j) {
                unsigned pk = (unsigned)f32_to_bf16(va[j]) | ((unsigned)f32_to_bf16(vc[j]) << 16);
                *reinterpret_cast<unsigned*>(&Bst[(n0 + j) * BSTR + k]) = pk;
            }
        }
    }
    __syncthreads();

    // ---- per-wave B fragments: all 8 K-steps held in registers ----
    const int ntile = wid & 1;           // which 16-col group
    const int mq    = wid >> 1;          // 0..3 -> 9 M-tiles each
    const int ncol  = ntile * 16 + (lane & 15);
    const int grp   = lane >> 4;         // 0/1

    Frag16 bfrag[8];
    {
        const unsigned short* bb = Bst + ncol * BSTR + grp * 16;  // K sequential per lane
        #pragma unroll
        for (int s = 0; s < 8; ++s) {
            const uint4* p = reinterpret_cast<const uint4*>(bb + s * 32);
            bfrag[s].u[0] = p[0];
            bfrag[s].u[1] = p[1];
        }
    }

    // ---- GEMM sweep over 9 M-tiles: preload A frags, then WMMA chain ----
    const int col = ntile * 16 + (lane & 15);
    for (int mt = 0; mt < 9; ++mt) {
        const int m0 = (mq * 9 + mt) * 16;
        const unsigned short* arow = Wb + (size_t)(m0 + (lane & 15)) * DCH;
        if (mt + 1 < 9)
            __builtin_prefetch(arow + 16 * DCH, 0, 3);   // near-scope global_prefetch
        Frag16 afr[8];
        #pragma unroll
        for (int s = 0; s < 8; ++s) {
            afr[s].u[0] = *reinterpret_cast<const uint4*>(arow + s * 32 + grp * 16);
            afr[s].u[1] = *reinterpret_cast<const uint4*>(arow + s * 32 + grp * 16 + 8);
        }
        v8f acc = {};
        #pragma unroll
        for (int s = 0; s < 8; ++s)
            acc = __builtin_amdgcn_wmma_f32_16x16x32_bf16(
                      false, afr[s].v, false, bfrag[s].v, (short)0, acc, false, false);
        const int rbase = m0 + 8 * grp;
        #pragma unroll
        for (int r = 0; r < 8; ++r)
            Res[(rbase + r) * NT + col] = acc[r];
    }
    __syncthreads();

    // ---- epilogue: softmax + histogram cutoff ----
    for (int i = tid; i < NBIN; i += 256) hist[i] = 0;
    __syncthreads();

    for (int tau = tid; tau < NT * NANCH; tau += 256) {
        const int n = tau & (NT - 1), a = tau >> 5;
        float x[NCLS], mx = -1e30f;
        #pragma unroll
        for (int c = 0; c < NCLS; ++c) {
            x[c] = Res[(a * NCLS + c) * NT + n] + cls_b[a * NCLS + c];
            mx = fmaxf(mx, x[c]);
        }
        float sum = 0.f;
        #pragma unroll
        for (int c = 0; c < NCLS; ++c) sum += __expf(x[c] - mx);
        const float inv = 1.0f / sum;
        #pragma unroll
        for (int c = 1; c < NCLS; ++c) {
            float sc = __expf(x[c] - mx) * inv;
            if (sc > SCORE_TH) {
                int bb = min(NBIN - 1, (int)(sc * (float)NBIN));
                atomicAdd(&hist[bb], 1);
            }
        }
    }
    __syncthreads();

    if (tid == 0) {
        int cum = 0, cut = 0;
        for (int bb = NBIN - 1; bb >= 0; --bb) {
            cum += hist[bb];
            if (cum >= TOPK) { cut = bb; break; }
        }
        s_cut = cut;
    }
    __syncthreads();

    // ---- emission pass: decode + write survivors ----
    for (int tau = tid; tau < NT * NANCH; tau += 256) {
        const int n = tau & (NT - 1), a = tau >> 5;
        const int l = l0 + n;
        float x[NCLS], mx = -1e30f;
        #pragma unroll
        for (int c = 0; c < NCLS; ++c) {
            x[c] = Res[(a * NCLS + c) * NT + n] + cls_b[a * NCLS + c];
            mx = fmaxf(mx, x[c]);
        }
        float sum = 0.f;
        #pragma unroll
        for (int c = 0; c < NCLS; ++c) sum += __expf(x[c] - mx);
        const float inv = 1.0f / sum;
        const float wa  = c_WA[a];
        const float ctr = (float)l * stride;
        #pragma unroll
        for (int c = 1; c < NCLS; ++c) {
            float sc = __expf(x[c] - mx) * inv;
            if (sc > SCORE_TH) {
                int bb = min(NBIN - 1, (int)(sc * (float)NBIN));
                if (bb >= s_cut) {
                    int slot = atomicAdd(&counters[b], 1);
                    if (slot < cap) {
                        int ro = BOX_OFF + a * 42 + 2 * c;
                        float dx = Res[ro * NT + n] + box_b[a * 42 + 2 * c];
                        float dw = fminf(Res[(ro + 1) * NT + n] + box_b[a * 42 + 2 * c + 1], CLIP_DW);
                        float pc = dx * wa + ctr;
                        float pw = __expf(dw) * wa;
                        float x0 = fminf(fmaxf(pc - 0.5f * pw, 0.f), imsz);
                        float x1 = fminf(fmaxf(pc + 0.5f * pw, 0.f), imsz);
                        float4 rec;
                        rec.x = sc; rec.y = x0; rec.z = x1;
                        rec.w = __int_as_float((l * NANCH + a) * 20 + (c - 1));
                        cand[(size_t)b * cap + slot] = rec;
                    }
                }
            }
        }
    }
}

// --------------------------------------------------------------- merge ----
// Phase 1: batch-level histogram -> cutoff for top-100.
// Phase 2: compact candidates >= cutoff into LDS pool (<=512).
// Phase 3: single wave32 does exact top-100 by shuffle-reduce (no barriers).
__global__ void merge_topk(const float4* __restrict__ cand,
                           const int* __restrict__ counters,
                           float* __restrict__ out,
                           int cap) {
    const int b = blockIdx.x;
    const int tid = threadIdx.x;
    const int n = min(counters[b], cap);
    const float4* cb = cand + (size_t)b * cap;

    __shared__ int    hist[NBIN];
    __shared__ int    s_cut, s_cnt;
    __shared__ float4 pool[POOLSZ];

    for (int i = tid; i < NBIN; i += 256) hist[i] = 0;
    if (tid == 0) s_cnt = 0;
    __syncthreads();

    for (int i = tid; i < n; i += 256) {
        float sc = cb[i].x;
        int bb = min(NBIN - 1, (int)(sc * (float)NBIN));
        atomicAdd(&hist[bb], 1);
    }
    __syncthreads();

    if (tid == 0) {
        int cum = 0, cut = 0;
        for (int bb = NBIN - 1; bb >= 0; --bb) {
            cum += hist[bb];
            if (cum >= TOPK) { cut = bb; break; }
        }
        s_cut = cut;
    }
    __syncthreads();

    for (int i = tid; i < n; i += 256) {
        float4 r = cb[i];
        int bb = min(NBIN - 1, (int)(r.x * (float)NBIN));
        if (bb >= s_cut) {
            int s = atomicAdd(&s_cnt, 1);
            if (s < POOLSZ) pool[s] = r;
        }
    }
    __syncthreads();

    if (tid < 32) {
        const int m = min(s_cnt, POOLSZ);
        const int lane = tid;
        unsigned taken = 0;                  // bit j -> pool[lane + 32*j] consumed
        for (int it = 0; it < TOPK; ++it) {
            float best = -1.f; int bfl = 0x7fffffff; int bj = -1;
            #pragma unroll 4
            for (int j = 0; j * 32 + lane < m; ++j) {
                if ((taken >> j) & 1u) continue;
                float4 r = pool[lane + 32 * j];
                int fi = __float_as_int(r.w);
                if (r.x > best || (r.x == best && fi < bfl)) { best = r.x; bfl = fi; bj = j; }
            }
            float rb = best; int rf = bfl; int rl = lane; int rj = bj;
            #pragma unroll
            for (int off = 16; off > 0; off >>= 1) {
                float ob = __shfl_xor(rb, off);
                int   of = __shfl_xor(rf, off);
                int   ol = __shfl_xor(rl, off);
                int   oj = __shfl_xor(rj, off);
                if (ob > rb || (ob == rb && of < rf)) { rb = ob; rf = of; rl = ol; rj = oj; }
            }
            if (lane == 0) {
                float* o = out + ((size_t)b * TOPK + it) * 3;
                if (rb >= 0.f && rj >= 0) {
                    float4 r = pool[rl + 32 * rj];
                    o[0] = r.y; o[1] = r.z; o[2] = r.x;
                } else {
                    o[0] = 0.f; o[1] = 0.f; o[2] = 0.f;
                }
            }
            if (lane == rl && rj >= 0 && rb >= 0.f) taken |= (1u << rj);
        }
    }
}

// -------------------------------------------------------------- launch ----
extern "C" void kernel_launch(void* const* d_in, const int* in_sizes, int n_in,
                              void* d_out, int out_size, void* d_ws, size_t ws_size,
                              hipStream_t stream) {
    const float* F     = (const float*)d_in[0];
    const float* cls_w = (const float*)d_in[1];
    const float* cls_b = (const float*)d_in[2];
    const float* box_w = (const float*)d_in[3];
    const float* box_b = (const float*)d_in[4];
    const int*   img   = (const int*)d_in[5];
    float* out = (float*)d_out;

    char* ws = (char*)d_ws;
    unsigned short* Wb = (unsigned short*)ws;                 // 576*256*2 = 294912 B
    int* counters = (int*)(ws + 294912);                      // 4 ints
    float4* cand  = (float4*)(ws + 295168);
    size_t avail  = (ws_size > 295168) ? (ws_size - 295168) : 0;
    int cap = (int)(avail / (4 * sizeof(float4)));
    if (cap > 262144) cap = 262144;
    if (cap < 1) cap = 1;

    const int smem = MTOT * NT * 4 + NT * BSTR * 2;           // 73728 + 16896 = 90624
    hipFuncSetAttribute((const void*)detect_main,
                        hipFuncAttributeMaxDynamicSharedMemorySize, smem);

    prep_weights<<<dim3(MTOT), dim3(DCH), 0, stream>>>(cls_w, box_w, Wb, counters);
    detect_main<<<dim3(L_POS / NT, 4), dim3(256), smem, stream>>>(
        F, Wb, cls_b, box_b, img, cand, counters, cap);
    merge_topk<<<dim3(4), dim3(256), 0, stream>>>(cand, counters, out, cap);
}
